// GAT_14147622273290
// MI455X (gfx1250) — matrix operand order repaired and checked by reading, not compiled, not run on hardware
//
#include <hip/hip_runtime.h>
#include <hip/hip_bf16.h>

// ---------------------------------------------------------------------------
// GAT (2-layer, PyG-style with self-loops) for MI455X / gfx1250.
//   - GEMMs via v_wmma_f32_16x16x32_bf16 (bf16 A/B, f32 accumulate),
//     B matrix staged in LDS (pad 272B rows -> conflict-free ds_load_b128),
//     all A-fragments preloaded in one clause (no global waits between WMMAs)
//   - edge softmax via uint-encoded atomicMax + f32 atomicAdd (L2-resident)
//   - scatter aggregation via f32 atomics (xl fits in 192MB L2)
// ---------------------------------------------------------------------------

typedef __bf16 bf16_t;
typedef __attribute__((ext_vector_type(16))) __bf16 v16bf;
typedef __attribute__((ext_vector_type(8)))  float  v8f;

union Frag16 { v16bf v; uint4 q[2]; };

__device__ __forceinline__ unsigned short f2bf(float f) {
    unsigned u = __float_as_uint(f);
    unsigned r = u + 0x7FFFu + ((u >> 16) & 1u);   // round-to-nearest-even
    return (unsigned short)(r >> 16);
}
// Monotone float<->uint mapping for atomicMax-based segment max.
__device__ __forceinline__ unsigned fenc(float f) {
    unsigned u = __float_as_uint(f);
    return (u & 0x80000000u) ? ~u : (u | 0x80000000u);
}
__device__ __forceinline__ float fdec(unsigned k) {
    unsigned u = (k & 0x80000000u) ? (k ^ 0x80000000u) : ~k;
    return __uint_as_float(u);
}
__device__ __forceinline__ float lrelu(float e) { return e > 0.f ? e : 0.2f * e; }

__device__ __forceinline__ void edge_sd(const int* __restrict__ ei, int E, int k,
                                        int& s, int& d) {
    if (k < E) { s = ei[k]; d = ei[E + k]; }
    else       { s = k - E; d = k - E; }           // self-loop
}

// ---------------- elementwise converts ----------------
__global__ void k_f2bf(const float* __restrict__ in, unsigned short* __restrict__ out, int n) {
    int t = blockIdx.x * blockDim.x + threadIdx.x;
    if (t < n) out[t] = f2bf(in[t]);
}
// W [K][Ncol] row-major -> Wt bf16 [Ncol][K] row-major (B operand wants contiguous K)
__global__ void k_wt(const float* __restrict__ W, unsigned short* __restrict__ Wt,
                     int K, int Ncol) {
    int t = blockIdx.x * blockDim.x + threadIdx.x;
    if (t >= K * Ncol) return;
    int k = t / Ncol, n = t % Ncol;
    Wt[n * K + k] = f2bf(W[t]);
}

// ---------------- WMMA GEMM: D[M][NT*16] = A[M][128](bf16) * B[128][NT*16](bf16)
// Bt is B transposed: [NT*16][128] bf16. One wave -> 16 rows x NT*16 cols.
// B is staged in LDS once per block (row stride padded to 272B: stride/4 = 68,
// 68 mod 64 = 4 -> lane l15 starts at bank 4*l15, conflict-free b128 reads).
template <int NT>
__global__ __launch_bounds__(256) void k_gemm(const unsigned short* __restrict__ A,
                                              const unsigned short* __restrict__ Bt,
                                              float* __restrict__ D, int M) {
    constexpr int K    = 128;
    constexpr int ROWB = K * 2 + 16;                 // 272B padded LDS row
    constexpr int Ncol = NT * 16;
    __shared__ unsigned char ldsB[Ncol * ROWB];

    // cooperative fill: Ncol rows x 256B, two threads per row (128B each)
    if (threadIdx.x < Ncol * 2) {
        int row = threadIdx.x >> 1, hf = threadIdx.x & 1;
        const uint4* src = (const uint4*)(Bt + (size_t)row * K + hf * 64);
        uint4* dst = (uint4*)(&ldsB[row * ROWB + hf * 128]);
#pragma unroll
        for (int q = 0; q < 8; ++q) dst[q] = src[q];
    }
    __syncthreads();                                 // all waves participate

    int wave = blockIdx.x * (blockDim.x >> 5) + (threadIdx.x >> 5);
    int RT = (M + 15) >> 4;
    if (wave >= RT) return;                          // after barrier: no deadlock
    int row0 = wave << 4;
    int ln   = threadIdx.x & 31;
    int half = ln >> 4;          // 0: lanes 0-15, 1: lanes 16-31
    int l15  = ln & 15;
    int rA = row0 + l15; if (rA > M - 1) rA = M - 1; // tail-safe (EXEC stays full)

    // Preload all 4 A-fragments in one clause (K/32 = 4).
    // A-frag: lane (half,l15) holds row M=l15, K in {kb..kb+7, kb+16..kb+23}, kb=8*half
    Frag16 a[4];
    const unsigned short* ap = A + (size_t)rA * K + half * 8;
#pragma unroll
    for (int kk = 0; kk < 4; ++kk) {
        a[kk].q[0] = *(const uint4*)(ap + kk * 32);
        a[kk].q[1] = *(const uint4*)(ap + kk * 32 + 16);
    }

    v8f acc[NT] = {};
#pragma unroll
    for (int kk = 0; kk < 4; ++kk) {
#pragma unroll
        for (int t = 0; t < NT; ++t) {
            // B-frag: lane holds column N=l15 of tile t, K in {kb..kb+7, kb+16..kb+23}, kb=8*half
            Frag16 b;
            const unsigned char* bp = &ldsB[(t * 16 + l15) * ROWB + kk * 64 + half * 16];
            b.q[0] = *(const uint4*)(bp);
            b.q[1] = *(const uint4*)(bp + 32);
            acc[t] = __builtin_amdgcn_wmma_f32_16x16x32_bf16(
                false, a[kk].v, false, b.v, (short)0, acc[t], false, false);
        }
    }
#pragma unroll
    for (int t = 0; t < NT; ++t) {
#pragma unroll
        for (int r = 0; r < 8; ++r) {
            int row = row0 + half * 8 + r;           // C/D layout: VGPR r -> M = r + 8*half
            if (row < M) D[(size_t)row * Ncol + t * 16 + l15] = acc[t][r];
        }
    }
}

// ---------------- layer-1 node kernel: e_src/e_dst, init m/s, zero h1 ----------
__global__ void k_node1(const float* __restrict__ xl, const float* __restrict__ as,
                        const float* __restrict__ ad, float* __restrict__ es,
                        float* __restrict__ ed, unsigned* __restrict__ m,
                        float* __restrict__ s, float* __restrict__ h1, int N) {
    int t = blockIdx.x * blockDim.x + threadIdx.x;
    if (t >= N * 8) return;
    int i = t >> 3, h = t & 7;
    const float* xr = xl + (size_t)i * 128 + h * 16;
    float a = 0.f, b = 0.f;
#pragma unroll
    for (int c = 0; c < 16; c += 4) {
        float4 v  = *(const float4*)(xr + c);
        float4 ws = *(const float4*)(as + h * 16 + c);
        float4 wd = *(const float4*)(ad + h * 16 + c);
        a += v.x * ws.x + v.y * ws.y + v.z * ws.z + v.w * ws.w;
        b += v.x * wd.x + v.y * wd.y + v.z * wd.z + v.w * wd.w;
    }
    es[t] = a; ed[t] = b; m[t] = 0u; s[t] = 0.f;     // fenc(-inf) > 0u, so 0 is max-identity
    float4 z = make_float4(0.f, 0.f, 0.f, 0.f);
    float* hr = h1 + (size_t)i * 128 + h * 16;
#pragma unroll
    for (int c = 0; c < 16; c += 4) *(float4*)(hr + c) = z;
}

// ---------------- layer-1 edge passes (thread per edge*head) ----------------
__global__ void k_edge_max1(const int* __restrict__ ei, int E, int Etot,
                            const float* __restrict__ es, const float* __restrict__ ed,
                            unsigned* __restrict__ m) {
    int t = blockIdx.x * blockDim.x + threadIdx.x;
    if (t >= Etot * 8) return;
    int k = t >> 3, h = t & 7, s, d;
    edge_sd(ei, E, k, s, d);
    float e = lrelu(es[(size_t)s * 8 + h] + ed[(size_t)d * 8 + h]);
    atomicMax(&m[(size_t)d * 8 + h], fenc(e));
}
__global__ void k_edge_sum1(const int* __restrict__ ei, int E, int Etot,
                            const float* __restrict__ es, const float* __restrict__ ed,
                            const unsigned* __restrict__ m, float* __restrict__ ssum) {
    int t = blockIdx.x * blockDim.x + threadIdx.x;
    if (t >= Etot * 8) return;
    int k = t >> 3, h = t & 7, s, d;
    edge_sd(ei, E, k, s, d);
    float e = lrelu(es[(size_t)s * 8 + h] + ed[(size_t)d * 8 + h]);
    atomicAdd(&ssum[(size_t)d * 8 + h], __expf(e - fdec(m[(size_t)d * 8 + h])));
}
// wave per edge: lane ln covers channels [4*ln, 4*ln+4), head h = ln>>2
__global__ void k_edge_agg1(const int* __restrict__ ei, int E, int Etot,
                            const float* __restrict__ es, const float* __restrict__ ed,
                            const unsigned* __restrict__ m, const float* __restrict__ ssum,
                            const float* __restrict__ xl, float* __restrict__ h1) {
    int wid = (blockIdx.x * blockDim.x + threadIdx.x) >> 5;
    if (wid >= Etot) return;
    int ln = threadIdx.x & 31, s, d;
    edge_sd(ei, E, wid, s, d);
    int h = ln >> 2;
    float e  = lrelu(es[(size_t)s * 8 + h] + ed[(size_t)d * 8 + h]);
    float mm = fdec(m[(size_t)d * 8 + h]);
    float alpha = __expf(e - mm) / (ssum[(size_t)d * 8 + h] + 1e-16f);
    float4 v = *(const float4*)(xl + (size_t)s * 128 + ln * 4);
    float* o = h1 + (size_t)d * 128 + ln * 4;
    atomicAdd(o + 0, alpha * v.x); atomicAdd(o + 1, alpha * v.y);
    atomicAdd(o + 2, alpha * v.z); atomicAdd(o + 3, alpha * v.w);
}

// ---------------- between layers: bias+ELU -> bf16, init layer-2 state --------
__global__ void k_node2a(const float* __restrict__ h1, const float* __restrict__ b1,
                         unsigned short* __restrict__ h1bf, float* __restrict__ out2,
                         unsigned* __restrict__ m2, float* __restrict__ s2, int N) {
    int t = blockIdx.x * blockDim.x + threadIdx.x;
    if (t >= N * 128) return;
    int j = t & 127, i = t >> 7;
    float v = h1[t] + b1[j];
    v = v > 0.f ? v : (__expf(v) - 1.f);             // ELU
    h1bf[t] = f2bf(v);
    if (j < 64) out2[(size_t)i * 64 + j] = 0.f;
    if (j == 0) { m2[i] = 0u; s2[i] = 0.f; }
}
__global__ void k_node2b(const float* __restrict__ h2, const float* __restrict__ as,
                         const float* __restrict__ ad, float* __restrict__ es,
                         float* __restrict__ ed, int N) {
    int i = blockIdx.x * blockDim.x + threadIdx.x;
    if (i >= N) return;
    const float* xr = h2 + (size_t)i * 64;
    float a = 0.f, b = 0.f;
#pragma unroll
    for (int c = 0; c < 64; c += 4) {
        float4 v  = *(const float4*)(xr + c);
        float4 ws = *(const float4*)(as + c);
        float4 wd = *(const float4*)(ad + c);
        a += v.x * ws.x + v.y * ws.y + v.z * ws.z + v.w * ws.w;
        b += v.x * wd.x + v.y * wd.y + v.z * wd.z + v.w * wd.w;
    }
    es[i] = a; ed[i] = b;
}

// ---------------- layer-2 edge passes (H=1) ----------------
__global__ void k_edge_max2(const int* __restrict__ ei, int E, int Etot,
                            const float* __restrict__ es, const float* __restrict__ ed,
                            unsigned* __restrict__ m) {
    int k = blockIdx.x * blockDim.x + threadIdx.x;
    if (k >= Etot) return;
    int s, d; edge_sd(ei, E, k, s, d);
    atomicMax(&m[d], fenc(lrelu(es[s] + ed[d])));
}
__global__ void k_edge_sum2(const int* __restrict__ ei, int E, int Etot,
                            const float* __restrict__ es, const float* __restrict__ ed,
                            const unsigned* __restrict__ m, float* __restrict__ ssum) {
    int k = blockIdx.x * blockDim.x + threadIdx.x;
    if (k >= Etot) return;
    int s, d; edge_sd(ei, E, k, s, d);
    float e = lrelu(es[s] + ed[d]);
    atomicAdd(&ssum[d], __expf(e - fdec(m[d])));
}
__global__ void k_edge_agg2(const int* __restrict__ ei, int E, int Etot,
                            const float* __restrict__ es, const float* __restrict__ ed,
                            const unsigned* __restrict__ m, const float* __restrict__ ssum,
                            const float* __restrict__ h2, float* __restrict__ out2) {
    int wid = (blockIdx.x * blockDim.x + threadIdx.x) >> 5;
    if (wid >= Etot) return;
    int ln = threadIdx.x & 31, s, d;
    edge_sd(ei, E, wid, s, d);
    float e = lrelu(es[s] + ed[d]);
    float alpha = __expf(e - fdec(m[d])) / (ssum[d] + 1e-16f);
    float2 v = *(const float2*)(h2 + (size_t)s * 64 + ln * 2);
    float* o = out2 + (size_t)d * 64 + ln * 2;
    atomicAdd(o + 0, alpha * v.x); atomicAdd(o + 1, alpha * v.y);
}

// ---------------- pooling ----------------
__global__ void k_init_pool(float* __restrict__ pool, float* __restrict__ cnt) {
    int t = blockIdx.x * blockDim.x + threadIdx.x;
    if (t < 256 * 64) pool[t] = 0.f;
    if (t < 256) cnt[t] = 0.f;
}
__global__ void k_pool(const float* __restrict__ out2, const float* __restrict__ b2,
                       const int* __restrict__ batch, float* __restrict__ pool,
                       float* __restrict__ cnt, int N) {
    int t = blockIdx.x * blockDim.x + threadIdx.x;
    if (t >= N * 64) return;
    int i = t >> 6, c = t & 63;
    int g = batch[i];
    atomicAdd(&pool[g * 64 + c], out2[t] + b2[c]);
    if (c == 0) atomicAdd(&cnt[g], 1.f);
}
__global__ void k_final(const float* __restrict__ pool, const float* __restrict__ cnt,
                        float* __restrict__ out) {
    int t = blockIdx.x * blockDim.x + threadIdx.x;
    if (t >= 256 * 64) return;
    out[t] = pool[t] / fmaxf(cnt[t >> 6], 1.f);
}

// ---------------------------------------------------------------------------
extern "C" void kernel_launch(void* const* d_in, const int* in_sizes, int n_in,
                              void* d_out, int out_size, void* d_ws, size_t ws_size,
                              hipStream_t stream) {
    const float* x   = (const float*)d_in[0];
    const float* W1  = (const float*)d_in[1];
    const float* as1 = (const float*)d_in[2];
    const float* ad1 = (const float*)d_in[3];
    const float* b1  = (const float*)d_in[4];
    const float* W2  = (const float*)d_in[5];
    const float* as2 = (const float*)d_in[6];
    const float* ad2 = (const float*)d_in[7];
    const float* b2  = (const float*)d_in[8];
    const int*   ei  = (const int*)d_in[9];
    const int*   bat = (const int*)d_in[10];
    float* out = (float*)d_out;

    const int Fin = 128;                 // per reference
    const int N    = in_sizes[0] / Fin;
    const int E    = in_sizes[9] / 2;
    const int Etot = E + N;

    // -------- workspace arena (lifetime-based aliasing, ~142 MB) --------
    char* base = (char*)d_ws;
    size_t off = 0;
    auto alloc = [&](size_t bytes) { size_t r = off; off = (off + bytes + 255) & ~(size_t)255; return r; };
    size_t o_xl   = alloc((size_t)N * 128 * 4);   // layer-1 projections; reused as out2
    size_t o_h1   = alloc((size_t)N * 128 * 4);   // layer-1 aggregate;  reused as h2
    size_t o_xbf  = alloc((size_t)N * 128 * 2);   // bf16 x;             reused as h1bf
    size_t o_es1  = alloc((size_t)N * 8 * 4);
    size_t o_ed1  = alloc((size_t)N * 8 * 4);
    size_t o_m1   = alloc((size_t)N * 8 * 4);
    size_t o_s1   = alloc((size_t)N * 8 * 4);
    size_t o_es2  = alloc((size_t)N * 4);
    size_t o_ed2  = alloc((size_t)N * 4);
    size_t o_m2   = alloc((size_t)N * 4);
    size_t o_s2   = alloc((size_t)N * 4);
    size_t o_w1t  = alloc(128 * 128 * 2);
    size_t o_w2t  = alloc(128 * 64 * 2);
    size_t o_pool = alloc(256 * 64 * 4);
    size_t o_cnt  = alloc(256 * 4);

    float*          xl   = (float*)(base + o_xl);
    float*          out2 = (float*)(base + o_xl);   // alias: xl dead after agg1
    float*          h1   = (float*)(base + o_h1);
    float*          h2   = (float*)(base + o_h1);   // alias: h1 dead after node2a
    unsigned short* xbf  = (unsigned short*)(base + o_xbf);
    unsigned short* h1bf = (unsigned short*)(base + o_xbf); // alias: xbf dead after gemm1
    float*    es1 = (float*)(base + o_es1);
    float*    ed1 = (float*)(base + o_ed1);
    unsigned* m1  = (unsigned*)(base + o_m1);
    float*    s1  = (float*)(base + o_s1);
    float*    es2 = (float*)(base + o_es2);
    float*    ed2 = (float*)(base + o_ed2);
    unsigned* m2  = (unsigned*)(base + o_m2);
    float*    s2  = (float*)(base + o_s2);
    unsigned short* W1t = (unsigned short*)(base + o_w1t);
    unsigned short* W2t = (unsigned short*)(base + o_w2t);
    float* pool = (float*)(base + o_pool);
    float* cnt  = (float*)(base + o_cnt);

    auto cdiv = [](long long a, long long b) { return (int)((a + b - 1) / b); };
    const int B = 256;
    const int RTb = cdiv((N + 15) / 16, 8);   // gemm blocks (8 waves/block)

    // ---- convert + transpose weights
    k_f2bf<<<cdiv((long long)N * 128, B), B, 0, stream>>>(x, xbf, N * 128);
    k_wt  <<<cdiv(128 * 128, B), B, 0, stream>>>(W1, W1t, 128, 128);
    k_wt  <<<cdiv(128 * 64,  B), B, 0, stream>>>(W2, W2t, 128, 64);
    // ---- layer 1
    k_gemm<8><<<RTb, B, 0, stream>>>(xbf, W1t, xl, N);
    k_node1<<<cdiv((long long)N * 8, B), B, 0, stream>>>(xl, as1, ad1, es1, ed1, m1, s1, h1, N);
    k_edge_max1<<<cdiv((long long)Etot * 8, B), B, 0, stream>>>(ei, E, Etot, es1, ed1, m1);
    k_edge_sum1<<<cdiv((long long)Etot * 8, B), B, 0, stream>>>(ei, E, Etot, es1, ed1, m1, s1);
    k_edge_agg1<<<cdiv((long long)Etot * 32, B), B, 0, stream>>>(ei, E, Etot, es1, ed1, m1, s1, xl, h1);
    // ---- layer 2
    k_init_pool<<<cdiv(256 * 64, B), B, 0, stream>>>(pool, cnt);
    k_node2a<<<cdiv((long long)N * 128, B), B, 0, stream>>>(h1, b1, h1bf, out2, m2, s2, N);
    k_gemm<4><<<RTb, B, 0, stream>>>(h1bf, W2t, h2, N);
    k_node2b<<<cdiv(N, B), B, 0, stream>>>(h2, as2, ad2, es2, ed2, N);
    k_edge_max2<<<cdiv(Etot, B), B, 0, stream>>>(ei, E, Etot, es2, ed2, m2);
    k_edge_sum2<<<cdiv(Etot, B), B, 0, stream>>>(ei, E, Etot, es2, ed2, m2, s2);
    k_edge_agg2<<<cdiv((long long)Etot * 32, B), B, 0, stream>>>(ei, E, Etot, es2, ed2, m2, s2, h2, out2);
    // ---- global mean pooling
    k_pool<<<cdiv((long long)N * 64, B), B, 0, stream>>>(out2, b2, bat, pool, cnt, N);
    k_final<<<cdiv(256 * 64, B), B, 0, stream>>>(pool, cnt, out);
}
